// MoELayer_14628658610469
// MI455X (gfx1250) — matrix operand order repaired and checked by reading, not compiled, not run on hardware
//
#include <hip/hip_runtime.h>
#include <hip/hip_bf16.h>

// ---------------- CDNA5 (gfx1250) MoE layer, bf16 WMMA path ----------------
// x:[4096,1024] f32 -> out:[4096,1024] f32
// shared SwiGLU + top-2-of-8 routed SwiGLU via grouped GEMM dispatch.
// Weights pre-transposed+converted to bf16; hot loop is 128-bit LDS loads
// feeding v_wmma_f32_16x16x32_bf16 with 2x2 register blocking (2 ds/wmma).

#define DIMD 1024
#define HIDD 1024
#define NEXP 8
#define NTOK 4096
#define MT   32          // rows per M-tile
#define MAXR 8448        // max padded routed rows (8192 + 8*32)
#define MAXTILES ((MAXR + NTOK) / MT)   // 392
#define LDA  1032        // padded LDS row stride (halfs) for A/H  (516 dw % 64 = 4)
#define LDB  72          // padded LDS row stride (halfs) for B    (36 dw  % 64 = 36)

typedef __attribute__((ext_vector_type(16))) __bf16 v16bf;
typedef __attribute__((ext_vector_type(8)))  float  v8f;

union FragU {
    v16bf v;
    unsigned short s[16];
    uint4 q[2];
};

static __device__ __forceinline__ unsigned short f32_to_bf16(float f) {
    union { float f; unsigned u; } x; x.f = f;
    unsigned r = x.u + 0x7FFFu + ((x.u >> 16) & 1u);   // round-to-nearest-even
    return (unsigned short)(r >> 16);
}

#define WMMA_BF16(A, B, C) __builtin_amdgcn_wmma_f32_16x16x32_bf16( \
        false, (A), false, (B), (short)0, (C), false, false)

// ---------------------------------------------------------------- convert ---
__global__ void convert_bf16_kernel(const float* __restrict__ src,
                                    unsigned short* __restrict__ dst, int n4) {
    int i = blockIdx.x * blockDim.x + threadIdx.x;
    if (i >= n4) return;
    float4 v = ((const float4*)src)[i];
    union { unsigned short s[4]; uint2 u; } o;
    o.s[0] = f32_to_bf16(v.x);
    o.s[1] = f32_to_bf16(v.y);
    o.s[2] = f32_to_bf16(v.z);
    o.s[3] = f32_to_bf16(v.w);
    ((uint2*)dst)[i] = o.u;
}

// ----------------------------------------------- transpose+convert weights --
// dst[n*1024 + k] = bf16(src[k*1024 + n]); blockIdx.z selects matrix.
__global__ __launch_bounds__(256) void transpose_bf16_kernel(
    const float* __restrict__ src, unsigned short* __restrict__ dst) {
    __shared__ float tile[32][33];
    const size_t MZ = (size_t)DIMD * HIDD;
    src += (size_t)blockIdx.z * MZ;
    dst += (size_t)blockIdx.z * MZ;
    int n0 = blockIdx.x * 32, k0 = blockIdx.y * 32;
    int tx = threadIdx.x & 31, ty = threadIdx.x >> 5;   // 32 x 8
#pragma unroll
    for (int i = 0; i < 4; ++i) {
        int kk = ty * 4 + i;
        tile[kk][tx] = src[(size_t)(k0 + kk) * 1024 + n0 + tx];
    }
    __syncthreads();
#pragma unroll
    for (int i = 0; i < 4; ++i) {
        int nn = ty * 4 + i;
        dst[(size_t)(n0 + nn) * 1024 + k0 + tx] = f32_to_bf16(tile[tx][nn]);
    }
}

// ------------------------------------------------------------------- init ---
__global__ void init_kernel(int* __restrict__ counts, int* __restrict__ rowid) {
    int i = blockIdx.x * blockDim.x + threadIdx.x;
    if (i < NEXP) counts[i] = 0;
    if (i < MAXR) rowid[i] = -1;
}

// ----------------------------------------------------------------- router ---
__global__ __launch_bounds__(256) void router_kernel(
    const float* __restrict__ x, const float* __restrict__ Wr,
    const float* __restrict__ loop_table, const int* __restrict__ loop_idx,
    float* __restrict__ tokw, int* __restrict__ tokexp, int* __restrict__ counts) {
    int w    = blockIdx.x * 8 + (threadIdx.x >> 5);
    int lane = threadIdx.x & 31;
    if (w >= NTOK) return;
    const float* xr = x + (size_t)w * DIMD;
    const float* le = loop_table + (size_t)loop_idx[0] * DIMD;
    float acc[8];
#pragma unroll
    for (int e = 0; e < 8; ++e) acc[e] = 0.f;
    for (int d = lane; d < DIMD; d += 32) {
        float xv = xr[d];
        float lv = le[d];
        const float4* w0 = (const float4*)(Wr + (size_t)d * 8);
        const float4* w1 = (const float4*)(Wr + (size_t)(d + DIMD) * 8);
        float4 a = w0[0], b = w0[1], c = w1[0], dd = w1[1];
        acc[0] += xv * a.x + lv * c.x;  acc[1] += xv * a.y + lv * c.y;
        acc[2] += xv * a.z + lv * c.z;  acc[3] += xv * a.w + lv * c.w;
        acc[4] += xv * b.x + lv * dd.x; acc[5] += xv * b.y + lv * dd.y;
        acc[6] += xv * b.z + lv * dd.z; acc[7] += xv * b.w + lv * dd.w;
    }
#pragma unroll
    for (int off = 16; off; off >>= 1)
#pragma unroll
        for (int e = 0; e < 8; ++e) acc[e] += __shfl_xor(acc[e], off);
    if (lane == 0) {
        float p[8];
#pragma unroll
        for (int e = 0; e < 8; ++e) p[e] = 1.f / (1.f + __expf(-acc[e]));
        int i0 = 0;
        for (int e = 1; e < 8; ++e) if (p[e] > p[i0]) i0 = e;
        int i1 = -1;
        for (int e = 0; e < 8; ++e)
            if (e != i0 && (i1 < 0 || p[e] > p[i1])) i1 = e;
        tokexp[2 * w]     = i0;   tokexp[2 * w + 1] = i1;
        tokw[2 * w]       = p[i0]; tokw[2 * w + 1]  = p[i1];
        atomicAdd(&counts[i0], 1);
        atomicAdd(&counts[i1], 1);
    }
}

// ------------------------------------------------------------------- scan ---
__global__ void scan_kernel(const int* __restrict__ counts,
                            int* __restrict__ offp, int* __restrict__ cursors) {
    if (blockIdx.x == 0 && threadIdx.x == 0) {
        int off = 0;
        for (int e = 0; e < NEXP; ++e) {
            offp[e] = off;
            cursors[e] = off;
            off += ((counts[e] + MT - 1) / MT) * MT;   // pad to tile
        }
        offp[NEXP] = off;   // Rp: total padded routed rows
    }
}

// ---------------------------------------------------------------- scatter ---
__global__ void scatter_kernel(const int* __restrict__ tokexp,
                               int* __restrict__ cursors,
                               int* __restrict__ rowid, int* __restrict__ tokslot) {
    int t = blockIdx.x * blockDim.x + threadIdx.x;
    if (t >= NTOK) return;
    for (int k = 0; k < 2; ++k) {
        int e = tokexp[2 * t + k];
        int r = atomicAdd(&cursors[e], 1);
        rowid[r] = t;
        tokslot[2 * t + k] = r;
    }
}

// -------------------------------------------------- fused expert GEMM tile --
// Per 32-row tile: H = silu(A@Wg)*(A@Wu) resident in LDS, then C = H@Wd.
// Shared expert is "expert 8". Routed rows -> R, shared rows -> out.
__global__ __launch_bounds__(256) void moe_tile_kernel(
    const unsigned short* __restrict__ Xb,
    const unsigned short* __restrict__ WgT,   // [9][HID n][DIM k] bf16
    const unsigned short* __restrict__ WuT,
    const unsigned short* __restrict__ WdT,   // [9][DIM n][HID k] bf16
    const int* __restrict__ rowid, const int* __restrict__ offp,
    float* __restrict__ R, float* __restrict__ out) {
    extern __shared__ __align__(16) unsigned char smem[];
    unsigned short* A  = (unsigned short*)smem;           // 32 x LDA
    unsigned short* H  = A  + MT * LDA;                   // 32 x LDA
    unsigned short* Bt = H  + MT * LDA;                   // 256 x LDB
    int* tokL = (int*)(Bt + 256 * LDB);                   // 32 ints

    int Rp   = offp[NEXP];
    int row0 = blockIdx.x * MT;
    if (row0 >= Rp + NTOK) return;
    int tid = threadIdx.x;
    bool isShared = (row0 >= Rp);
    int e = NEXP;
    if (!isShared) { e = 0; while (row0 >= offp[e + 1]) ++e; }

    if (tid < MT)
        tokL[tid] = isShared ? (row0 - Rp + tid) : rowid[row0 + tid];
    __syncthreads();

    // --- gather A rows (bf16) into LDS: 8 threads/row, 16B vectors ---
    {
        int m = tid >> 3, part = tid & 7;
        int tk = tokL[m];
        const uint4* src = (const uint4*)(Xb + (size_t)(tk < 0 ? 0 : tk) * DIMD);
        uint4* dst = (uint4*)(A + m * LDA);
        uint4 z = make_uint4(0, 0, 0, 0);
#pragma unroll
        for (int i = 0; i < 16; ++i)
            dst[part * 16 + i] = (tk >= 0) ? src[part * 16 + i] : z;
    }
    __syncthreads();

    const unsigned short* Wg = WgT + (size_t)e * DIMD * HIDD;
    const unsigned short* Wu = WuT + (size_t)e * DIMD * HIDD;
    const unsigned short* Wd = WdT + (size_t)e * HIDD * DIMD;

    int wid  = tid >> 5, lane = tid & 31;
    int lmod = lane & 15;
    int kbaseA = (lane < 16) ? 0 : 8;    // A frag: K 0-7/16-23 vs 8-15/24-31
    int kbaseB = (lane < 16) ? 0 : 8;    // B frag: K 0-7/16-23 vs 8-15/24-31
    int hi   = lane >> 4;

    // ------- stage 1: H = silu(A@Wg)*(A@Wu); N-chunk 128, wave = 2M x 1N ----
    // Bt split: rows 0..127 gate, rows 128..255 up.
    unsigned short* Btg = Bt;
    unsigned short* Btu = Bt + 128 * LDB;
    for (int nc = 0; nc < HIDD / 128; ++nc) {
        v8f accg[2] = {{}, {}}, accu[2] = {{}, {}};
        int nSub = wid * 16;
        for (int k0 = 0; k0 < DIMD; k0 += 64) {
            __syncthreads();
            {   // stage B^T tiles: 256 threads -> 256 rows (128 g + 128 u)
                int rrow = tid & 127;
                const unsigned short* srcW = (tid < 128) ? Wg : Wu;
                unsigned short* dstB = (tid < 128) ? Btg : Btu;
                const uint4* s4 = (const uint4*)(srcW + (size_t)(nc * 128 + rrow) * DIMD + k0);
                uint4* d4 = (uint4*)(dstB + rrow * LDB);
#pragma unroll
                for (int j = 0; j < 4; ++j) d4[j] = s4[j];
                if (k0 + 64 < DIMD)
                    __builtin_prefetch(srcW + (size_t)(nc * 128 + rrow) * DIMD + k0 + 64, 0, 1);
            }
            __syncthreads();
#pragma unroll
            for (int kk = 0; kk < 64; kk += 32) {
                FragU a0, a1, bg, bu;
                const unsigned short* ar0 = A + lmod * LDA + k0 + kk + kbaseA;
                const unsigned short* ar1 = A + (16 + lmod) * LDA + k0 + kk + kbaseA;
                a0.q[0] = *(const uint4*)(ar0);  a0.q[1] = *(const uint4*)(ar0 + 16);
                a1.q[0] = *(const uint4*)(ar1);  a1.q[1] = *(const uint4*)(ar1 + 16);
                const unsigned short* bgr = Btg + (nSub + lmod) * LDB + kk + kbaseB;
                const unsigned short* bur = Btu + (nSub + lmod) * LDB + kk + kbaseB;
                bg.q[0] = *(const uint4*)(bgr);  bg.q[1] = *(const uint4*)(bgr + 16);
                bu.q[0] = *(const uint4*)(bur);  bu.q[1] = *(const uint4*)(bur + 16);
                accg[0] = WMMA_BF16(a0.v, bg.v, accg[0]);
                accg[1] = WMMA_BF16(a1.v, bg.v, accg[1]);
                accu[0] = WMMA_BF16(a0.v, bu.v, accu[0]);
                accu[1] = WMMA_BF16(a1.v, bu.v, accu[1]);
            }
        }
#pragma unroll
        for (int mi = 0; mi < 2; ++mi)
#pragma unroll
            for (int r = 0; r < 8; ++r) {   // silu(g)*u -> H (bf16)
                float g = accg[mi][r], u = accu[mi][r];
                float hval = (g / (1.f + __expf(-g))) * u;
                int m = mi * 16 + r + 8 * hi;
                H[m * LDA + nc * 128 + nSub + lmod] = f32_to_bf16(hval);
            }
    }
    __syncthreads();

    // ------- stage 2: C = H @ Wd; N-chunk 256, wave = 2M x 2N ---------------
    for (int nc = 0; nc < DIMD / 256; ++nc) {
        v8f acc[2][2] = {{{}, {}}, {{}, {}}};   // [mi][ni]
        int nSub = wid * 32;
        for (int k0 = 0; k0 < HIDD; k0 += 64) {
            __syncthreads();
            {   // stage B^T tile: 256 threads -> 256 rows
                const uint4* s4 = (const uint4*)(Wd + (size_t)(nc * 256 + tid) * HIDD + k0);
                uint4* d4 = (uint4*)(Bt + tid * LDB);
#pragma unroll
                for (int j = 0; j < 4; ++j) d4[j] = s4[j];
                if (k0 + 64 < HIDD)
                    __builtin_prefetch(Wd + (size_t)(nc * 256 + tid) * HIDD + k0 + 64, 0, 1);
            }
            __syncthreads();
#pragma unroll
            for (int kk = 0; kk < 64; kk += 32) {
                FragU a0, a1, b0, b1;
                const unsigned short* ar0 = H + lmod * LDA + k0 + kk + kbaseA;
                const unsigned short* ar1 = H + (16 + lmod) * LDA + k0 + kk + kbaseA;
                a0.q[0] = *(const uint4*)(ar0);  a0.q[1] = *(const uint4*)(ar0 + 16);
                a1.q[0] = *(const uint4*)(ar1);  a1.q[1] = *(const uint4*)(ar1 + 16);
                const unsigned short* br0 = Bt + (nSub + lmod) * LDB + kk + kbaseB;
                const unsigned short* br1 = Bt + (nSub + 16 + lmod) * LDB + kk + kbaseB;
                b0.q[0] = *(const uint4*)(br0);  b0.q[1] = *(const uint4*)(br0 + 16);
                b1.q[0] = *(const uint4*)(br1);  b1.q[1] = *(const uint4*)(br1 + 16);
                acc[0][0] = WMMA_BF16(a0.v, b0.v, acc[0][0]);
                acc[1][0] = WMMA_BF16(a1.v, b0.v, acc[1][0]);
                acc[0][1] = WMMA_BF16(a0.v, b1.v, acc[0][1]);
                acc[1][1] = WMMA_BF16(a1.v, b1.v, acc[1][1]);
            }
        }
#pragma unroll
        for (int mi = 0; mi < 2; ++mi)
#pragma unroll
            for (int r = 0; r < 8; ++r) {
                int m = mi * 16 + r + 8 * hi;
                int tk = tokL[m];
                if (tk < 0) continue;
                float* base = isShared ? (out + (size_t)tk * DIMD)
                                       : (R + (size_t)(row0 + m) * DIMD);
#pragma unroll
                for (int ni = 0; ni < 2; ++ni)
                    base[nc * 256 + nSub + ni * 16 + lmod] = acc[mi][ni][r];
            }
    }
}

// ---------------------------------------------------------------- combine ---
__global__ void combine_kernel(float* __restrict__ out, const float* __restrict__ R,
                               const int* __restrict__ tokslot,
                               const float* __restrict__ tokw) {
    int i = blockIdx.x * blockDim.x + threadIdx.x;   // over NTOK*DIMD/4
    if (i >= NTOK * (DIMD / 4)) return;
    int token = i >> 8;        // 256 float4 per row
    int c4    = i & 255;
    int s0 = tokslot[2 * token], s1 = tokslot[2 * token + 1];
    float w0 = tokw[2 * token], w1 = tokw[2 * token + 1];
    float4 r0 = ((const float4*)(R + (size_t)s0 * DIMD))[c4];
    float4 r1 = ((const float4*)(R + (size_t)s1 * DIMD))[c4];
    float4* o = (float4*)(out + (size_t)token * DIMD) + c4;
    float4 v = *o;
    v.x += w0 * r0.x + w1 * r1.x;
    v.y += w0 * r0.y + w1 * r1.y;
    v.z += w0 * r0.z + w1 * r1.z;
    v.w += w0 * r0.w + w1 * r1.w;
    *o = v;
}

// ------------------------------------------------------------------ launch --
extern "C" void kernel_launch(void* const* d_in, const int* in_sizes, int n_in,
                              void* d_out, int out_size, void* d_ws, size_t ws_size,
                              hipStream_t stream) {
    (void)in_sizes; (void)n_in; (void)out_size; (void)ws_size;
    const float* x  = (const float*)d_in[0];
    const float* sg = (const float*)d_in[1];
    const float* su = (const float*)d_in[2];
    const float* sd = (const float*)d_in[3];
    const float* Wg = (const float*)d_in[4];
    const float* Wu = (const float*)d_in[5];
    const float* Wd = (const float*)d_in[6];
    const float* Wr = (const float*)d_in[7];
    const float* lt = (const float*)d_in[8];
    const int* loop_idx = (const int*)d_in[9];
    float* out = (float*)d_out;

    // ---- workspace layout ----
    unsigned char* p = (unsigned char*)d_ws;
    size_t o = 0;
    auto take = [&](size_t bytes) { void* r = p + o; o = (o + bytes + 255) & ~(size_t)255; return r; };
    const size_t MATB = (size_t)DIMD * HIDD * sizeof(unsigned short);   // 2 MB
    unsigned short* WgT = (unsigned short*)take(9 * MATB);  // W^T: Wg[0..7], sg
    unsigned short* WuT = (unsigned short*)take(9 * MATB);
    unsigned short* WdT = (unsigned short*)take(9 * MATB);
    unsigned short* Xb  = (unsigned short*)take((size_t)NTOK * DIMD * 2);
    float* R       = (float*)take((size_t)MAXR * DIMD * sizeof(float));
    int* counts    = (int*)take(NEXP * sizeof(int));
    int* offp      = (int*)take((NEXP + 1) * sizeof(int));
    int* cursors   = (int*)take(NEXP * sizeof(int));
    int* rowid     = (int*)take(MAXR * sizeof(int));
    int* tokexp    = (int*)take(2 * NTOK * sizeof(int));
    int* tokslot   = (int*)take(2 * NTOK * sizeof(int));
    float* tokw    = (float*)take(2 * NTOK * sizeof(float));

    // ---- x: fp32 -> bf16 (row-major kept) ----
    {
        int n4 = (int)((size_t)NTOK * DIMD / 4);
        convert_bf16_kernel<<<(n4 + 255) / 256, 256, 0, stream>>>(x, Xb, n4);
    }
    // ---- weights: fp32 -> bf16 with transpose ----
    auto trans = [&](const float* s, unsigned short* d, int nmat) {
        dim3 grid(32, 32, nmat);
        transpose_bf16_kernel<<<grid, 256, 0, stream>>>(s, d);
    };
    const size_t M1 = (size_t)DIMD * HIDD;
    trans(Wg, WgT, 8);  trans(sg, WgT + 8 * M1, 1);
    trans(Wu, WuT, 8);  trans(su, WuT + 8 * M1, 1);
    trans(Wd, WdT, 8);  trans(sd, WdT + 8 * M1, 1);

    init_kernel<<<(MAXR + 255) / 256, 256, 0, stream>>>(counts, rowid);
    router_kernel<<<NTOK / 8, 256, 0, stream>>>(x, Wr, lt, loop_idx,
                                                tokw, tokexp, counts);
    scan_kernel<<<1, 32, 0, stream>>>(counts, offp, cursors);
    scatter_kernel<<<(NTOK + 255) / 256, 256, 0, stream>>>(tokexp, cursors,
                                                           rowid, tokslot);

    size_t smem = (size_t)MT * LDA * 2 * 2    // A + H
                + (size_t)256 * LDB * 2      // Bt
                + MT * sizeof(int);          // tokL
    moe_tile_kernel<<<MAXTILES, 256, smem, stream>>>(
        Xb, WgT, WuT, WdT, rowid, offp, R, out);

    combine_kernel<<<(NTOK * (DIMD / 4) + 255) / 256, 256, 0, stream>>>(
        out, R, tokslot, tokw);
}
